// MultiHeadAttention_87273735454944
// MI455X (gfx1250) — matrix operand order repaired and checked by reading, not compiled, not run on hardware
//
#include <hip/hip_runtime.h>
#include <hip/hip_bf16.h>
#include <math.h>

typedef __attribute__((ext_vector_type(16))) _Float16 v16h;
typedef __attribute__((ext_vector_type(8)))  _Float16 v8h;
typedef __attribute__((ext_vector_type(4)))  _Float16 v4h;
typedef __attribute__((ext_vector_type(8)))  float    v8f;
typedef __attribute__((ext_vector_type(4)))  float    v4f;
typedef __attribute__((ext_vector_type(4)))  unsigned v4u;

#define DIM   768
#define NHEAD 16
#define DK    48
#define DKP   64
#define SEQ   2048
#define BATCH 4
#define MROWS (BATCH*SEQ)   // 8192

// ---------------- WMMA + fragment helpers ----------------

__device__ __forceinline__ v8f wmma_f16(v16h a, v16h b, v8f c) {
  // D(f32 16x16) = A(f16 16x32) * B(f16 32x16) + C
  return __builtin_amdgcn_wmma_f32_16x16x32_f16(false, a, false, b, (short)0, c,
                                                false, false);
}

// A-fragment per ISA layout: lane<16 holds row m, halves [base..base+7] and
// [base+16..base+23]; lane>=16 gets base+=8 (caller folds that into p).
__device__ __forceinline__ v16h ld_gap16(const _Float16* p) {
  v8h lo = *reinterpret_cast<const v8h*>(p);
  v8h hi = *reinterpret_cast<const v8h*>(p + 16);
  return __builtin_shufflevector(lo, hi, 0,1,2,3,4,5,6,7,8,9,10,11,12,13,14,15);
}
// B-fragment: lane holds 16 consecutive k values of its column (row of the
// pre-transposed operand): contiguous 32 bytes.
__device__ __forceinline__ v16h ld_seq16(const _Float16* p) {
  v8h lo = *reinterpret_cast<const v8h*>(p);
  v8h hi = *reinterpret_cast<const v8h*>(p + 8);
  return __builtin_shufflevector(lo, hi, 0,1,2,3,4,5,6,7,8,9,10,11,12,13,14,15);
}
__device__ __forceinline__ unsigned pk2(float a, float b) {
  auto t = __builtin_amdgcn_cvt_pkrtz(a, b);   // __fp16 ext_vector(2)
  return __builtin_bit_cast(unsigned, t);
}

// ---------------- prep: f32->f16 convert / transpose ----------------

__global__ void cvt_f32_f16(const float* __restrict__ in,
                            _Float16* __restrict__ out, int n4) {
  int i = blockIdx.x * blockDim.x + threadIdx.x;
  if (i >= n4) return;
  v4f v = reinterpret_cast<const v4f*>(in)[i];
  v4h o;
  o[0] = (_Float16)v[0]; o[1] = (_Float16)v[1];
  o[2] = (_Float16)v[2]; o[3] = (_Float16)v[3];
  reinterpret_cast<v4h*>(out)[i] = o;
}

__global__ void cvt_w_transpose(const float* __restrict__ W,
                                _Float16* __restrict__ Wt) {
  int i = blockIdx.x * blockDim.x + threadIdx.x;   // coalesced write of Wt[n][k]
  if (i >= DIM * DIM) return;
  int n = i / DIM;
  int k = i - n * DIM;
  Wt[i] = (_Float16)W[(size_t)k * DIM + n];        // Wt[n][k] = W[k][n]
}

// ---------------- QKV projection GEMM (wave = 16x64 tile) ----------------
// mode 0: Q -> padded [b,h,s,DKP] * scale ; mode 1: K -> padded ; mode 2: V^T [b,h,dk,s]

__global__ void qkv_gemm_kernel(const _Float16* __restrict__ X,
                                const _Float16* __restrict__ Wt,
                                const float* __restrict__ bias,
                                _Float16* __restrict__ out,
                                int mode, float scale) {
  const int lane = threadIdx.x & 31;
  const int wave = threadIdx.x >> 5;
  const int l16  = lane & 15;
  const bool hiL = lane >= 16;
  const int aoff = hiL ? 8 : 0;
  const int boff = hiL ? 16 : 0;
  const int m0 = blockIdx.x * 16;
  const int nb = blockIdx.y * 256 + wave * 64;

  const _Float16* arow = X + (size_t)(m0 + l16) * DIM;
  v8f c[4] = {{}, {}, {}, {}};
  for (int k = 0; k < DIM; k += 32) {
    v16h a = ld_seq16(arow + k + boff);
#pragma unroll
    for (int t = 0; t < 4; ++t) {
      const _Float16* brow = Wt + (size_t)(nb + t * 16 + l16) * DIM + k + boff;
      c[t] = wmma_f16(a, ld_seq16(brow), c[t]);
    }
  }
  const int bb = m0 >> 11;                 // batch index (SEQ=2048)
  const int s0 = (m0 & (SEQ - 1)) + (hiL ? 8 : 0);
#pragma unroll
  for (int t = 0; t < 4; ++t) {
    const int n  = nb + t * 16 + l16;      // output feature
    const float bn = bias[n];
    const int hh = n / DK;
    const int j  = n - hh * DK;
    if (mode <= 1) {
      // padded [b,h,s,DKP]; rows m are s (stride DKP)
      _Float16* dst = out + ((size_t)(bb * NHEAD + hh) * SEQ + s0) * DKP + j;
#pragma unroll
      for (int g = 0; g < 8; ++g)
        dst[(size_t)g * DKP] = (_Float16)((c[t][g] + bn) * scale);
    } else {
      // V^T [b,h,dk,s]: 8 consecutive s values -> one b128 store
      _Float16* dst = out + ((size_t)(bb * NHEAD + hh) * DK + j) * SEQ + s0;
      v4u o;
#pragma unroll
      for (int i = 0; i < 4; ++i)
        o[i] = pk2(c[t][2 * i] + bn, c[t][2 * i + 1] + bn);
      *reinterpret_cast<v4u*>(dst) = o;
    }
  }
}

// ---------------- flash attention (wave = 16 q rows, streams 32 keys/iter) --

__global__ void attn_kernel(const _Float16* __restrict__ Qp,
                            const _Float16* __restrict__ Kp,
                            const _Float16* __restrict__ Vt,
                            _Float16* __restrict__ Ah) {
  const int lane = threadIdx.x & 31;
  const int wave = threadIdx.x >> 5;
  const int l16  = lane & 15;
  const bool hiL = lane >= 16;
  const int aoff = hiL ? 8 : 0;
  const int boff = hiL ? 16 : 0;
  const int bh = blockIdx.x;             // b*16+h
  const int b  = bh >> 4;
  const int h  = bh & 15;
  const int q0 = (blockIdx.y * 4 + wave) * 16;

  const _Float16* Qb = Qp + (size_t)bh * SEQ * DKP;
  const _Float16* Kb = Kp + (size_t)bh * SEQ * DKP;
  const _Float16* Vb = Vt + (size_t)bh * DK * SEQ;

  // Q as B-fragment of S^T = K*Q^T (lane = q column, contiguous dk run)
  const _Float16* qrow = Qb + (size_t)(q0 + l16) * DKP;
  const v16h bq0 = ld_seq16(qrow + boff);        // dk 0..31
  const v16h bq1 = ld_seq16(qrow + 32 + boff);   // dk 32..63 (zero padded)

  v8f acc[3] = {{}, {}, {}};                     // O^T, dk tiles 0..2
  float m_run = -__builtin_inff();
  float l_run = 0.0f;

  for (int kb = 0; kb < SEQ; kb += 32) {
    // scores^T: two 16(key)x16(q) tiles, K-dim = DKP = 2 WMMAs each
    const _Float16* kr0 = Kb + (size_t)(kb + l16) * DKP;
    const _Float16* kr1 = Kb + (size_t)(kb + 16 + l16) * DKP;
    v8f z = {};
    v8f st0 = wmma_f16(ld_seq16(kr0 + boff), bq0, z);
    st0 = wmma_f16(ld_seq16(kr0 + 32 + boff), bq1, st0);
    v8f st1 = wmma_f16(ld_seq16(kr1 + boff), bq0, z);
    st1 = wmma_f16(ld_seq16(kr1 + 32 + boff), bq1, st1);
    // lane now holds 16 key-scores for q = q0 + l16 (8 in st0, 8 in st1);
    // the other 16 keys of this block live in the partner lane (xor 16).

    float mx = -__builtin_inff();
#pragma unroll
    for (int g = 0; g < 8; ++g) { mx = fmaxf(mx, st0[g]); mx = fmaxf(mx, st1[g]); }
    mx = fmaxf(mx, __shfl_xor(mx, 16));
    const float m_new = fmaxf(m_run, mx);
    const float alpha = __expf(m_run - m_new);   // first iter: exp(-inf)=0
    m_run = m_new;

    float p0[8], p1[8];
    float rs = 0.0f;
#pragma unroll
    for (int g = 0; g < 8; ++g) {
      p0[g] = __expf(st0[g] - m_new);
      p1[g] = __expf(st1[g] - m_new);
      rs += p0[g] + p1[g];
    }
    rs += __shfl_xor(rs, 16);
    l_run = l_run * alpha + rs;

#pragma unroll
    for (int t = 0; t < 3; ++t)
#pragma unroll
      for (int g = 0; g < 8; ++g) acc[t][g] *= alpha;

    // Build P^T B-fragment (32 keys x 16 q) from own + partner exp values.
    v4u e0, e1;
#pragma unroll
    for (int i = 0; i < 4; ++i) {
      e0[i] = pk2(p0[2 * i], p0[2 * i + 1]);
      e1[i] = pk2(p1[2 * i], p1[2 * i + 1]);
    }
    v4u f0, f1;
#pragma unroll
    for (int i = 0; i < 4; ++i) {
      f0[i] = __shfl_xor(e0[i], 16);
      f1[i] = __shfl_xor(e1[i], 16);
    }
    const v4u blo = hiL ? f1 : e0;   // keys k..k+7   (lo lanes) / k+16..k+23 (hi lanes)
    const v4u bhi = hiL ? e1 : f0;   // keys k+8..k+15 (lo)      / k+24..k+31 (hi)
    const v16h bp = __builtin_shufflevector(
        __builtin_bit_cast(v8h, blo), __builtin_bit_cast(v8h, bhi),
        0,1,2,3,4,5,6,7,8,9,10,11,12,13,14,15);

    // O^T += V^T * P^T  (A rows contiguous in Vt)
#pragma unroll
    for (int t = 0; t < 3; ++t) {
      const _Float16* vrow = Vb + (size_t)(t * 16 + l16) * SEQ + kb;
      acc[t] = wmma_f16(ld_seq16(vrow + boff), bp, acc[t]);
    }
  }

  const float inv_l = 1.0f / l_run;
  const int s = q0 + l16;
  _Float16* orow = Ah + ((size_t)b * SEQ + s) * DIM + h * DK;
#pragma unroll
  for (int t = 0; t < 3; ++t) {
    v4u o;
#pragma unroll
    for (int i = 0; i < 4; ++i)
      o[i] = pk2(acc[t][2 * i] * inv_l, acc[t][2 * i + 1] * inv_l);
    *reinterpret_cast<v4u*>(orow + t * 16 + aoff) = o;   // 8 contiguous dk feats
  }
}

// ---------------- output projection GEMM (f32 out + bias) ----------------

__global__ void oproj_kernel(const _Float16* __restrict__ A,
                             const _Float16* __restrict__ Wt,
                             const float* __restrict__ bias,
                             float* __restrict__ out) {
  const int lane = threadIdx.x & 31;
  const int wave = threadIdx.x >> 5;
  const int l16  = lane & 15;
  const bool hiL = lane >= 16;
  const int aoff = hiL ? 8 : 0;
  const int boff = hiL ? 16 : 0;
  const int m0 = blockIdx.x * 16;
  const int nb = blockIdx.y * 256 + wave * 64;

  const _Float16* arow = A + (size_t)(m0 + l16) * DIM;
  v8f c[4] = {{}, {}, {}, {}};
  for (int k = 0; k < DIM; k += 32) {
    v16h a = ld_seq16(arow + k + boff);
#pragma unroll
    for (int t = 0; t < 4; ++t) {
      const _Float16* brow = Wt + (size_t)(nb + t * 16 + l16) * DIM + k + boff;
      c[t] = wmma_f16(a, ld_seq16(brow), c[t]);
    }
  }
  const int row0 = m0 + (hiL ? 8 : 0);
#pragma unroll
  for (int t = 0; t < 4; ++t) {
    const int n = nb + t * 16 + l16;
    const float bn = bias[n];
#pragma unroll
    for (int g = 0; g < 8; ++g)
      out[(size_t)(row0 + g) * DIM + n] = c[t][g] + bn;
  }
}

// ---------------- host launch ----------------

extern "C" void kernel_launch(void* const* d_in, const int* in_sizes, int n_in,
                              void* d_out, int out_size, void* d_ws, size_t ws_size,
                              hipStream_t stream) {
  (void)in_sizes; (void)n_in; (void)out_size; (void)ws_size;
  const float* x  = (const float*)d_in[0];
  const float* Wq = (const float*)d_in[1];
  const float* bq = (const float*)d_in[2];
  const float* Wk = (const float*)d_in[3];
  const float* bk = (const float*)d_in[4];
  const float* Wv = (const float*)d_in[5];
  const float* bv = (const float*)d_in[6];
  const float* Wo = (const float*)d_in[7];
  const float* bo = (const float*)d_in[8];

  char* p = (char*)d_ws;
  _Float16* Xh  = (_Float16*)p; p += (size_t)MROWS * DIM * 2;
  _Float16* WqT = (_Float16*)p; p += (size_t)DIM * DIM * 2;
  _Float16* WkT = (_Float16*)p; p += (size_t)DIM * DIM * 2;
  _Float16* WvT = (_Float16*)p; p += (size_t)DIM * DIM * 2;
  _Float16* WoT = (_Float16*)p; p += (size_t)DIM * DIM * 2;
  _Float16* Qp  = (_Float16*)p; p += (size_t)BATCH * NHEAD * SEQ * DKP * 2;
  _Float16* Kp  = (_Float16*)p; p += (size_t)BATCH * NHEAD * SEQ * DKP * 2;
  _Float16* Vt  = (_Float16*)p; p += (size_t)BATCH * NHEAD * DK * SEQ * 2;
  _Float16* Ah  = Xh;  // reuse: Xh fully consumed by the QKV GEMMs (stream order)

  const int n4 = MROWS * DIM / 4;
  cvt_f32_f16<<<(n4 + 255) / 256, 256, 0, stream>>>(x, Xh, n4);
  const int nw = (DIM * DIM + 255) / 256;
  cvt_w_transpose<<<nw, 256, 0, stream>>>(Wq, WqT);
  cvt_w_transpose<<<nw, 256, 0, stream>>>(Wk, WkT);
  cvt_w_transpose<<<nw, 256, 0, stream>>>(Wv, WvT);
  cvt_w_transpose<<<nw, 256, 0, stream>>>(Wo, WoT);

  // zero the dk->DKP pad columns of Q/K (kept zero each call: deterministic)
  const size_t qk_bytes = (size_t)BATCH * NHEAD * SEQ * DKP * 2;
  (void)hipMemsetAsync(Qp, 0, qk_bytes, stream);
  (void)hipMemsetAsync(Kp, 0, qk_bytes, stream);

  const float qscale = 0.14433756729740643f;  // 1/sqrt(48), folded into Q
  dim3 gg(MROWS / 16, 3);
  qkv_gemm_kernel<<<gg, 128, 0, stream>>>(Xh, WqT, bq, Qp, 0, qscale);
  qkv_gemm_kernel<<<gg, 128, 0, stream>>>(Xh, WkT, bk, Kp, 1, 1.0f);
  qkv_gemm_kernel<<<gg, 128, 0, stream>>>(Xh, WvT, bv, Vt, 2, 1.0f);

  attn_kernel<<<dim3(BATCH * NHEAD, SEQ / 64), 128, 0, stream>>>(Qp, Kp, Vt, Ah);

  oproj_kernel<<<gg, 128, 0, stream>>>(Ah, WoT, bo, (float*)d_out);
}
